// TemporalGraphTransformer_7876970021013
// MI455X (gfx1250) — matrix operand order repaired and checked by reading, not compiled, not run on hardware
//
#include <hip/hip_runtime.h>
#include <hip/hip_bf16.h>
#include <math.h>

// ---------------------------------------------------------------------------
// GATv2 layer (B=4, N=10000, E=160000, Cin=256, H=4, C=128) for gfx1250.
// GEMMs: v_wmma_f32_16x16x32_bf16, register-tiled 16x64 per wave, B tile
// staged in LDS via async global->LDS copies. Softmax via f32 atomics.
// ---------------------------------------------------------------------------

typedef __attribute__((ext_vector_type(16))) __bf16 v16bf;
typedef __attribute__((ext_vector_type(8)))  float  v8f;
typedef int v4i __attribute__((vector_size(16)));   // matches async-LDS builtin

#define NEG_SLOPE 0.2f
#define LDSROW 40   // 64-col B tile, rows padded to 40 bf16 (80B) for banks

#if __has_builtin(__builtin_amdgcn_global_load_async_to_lds_b128) && \
    __has_builtin(__builtin_amdgcn_s_wait_asynccnt)
#define HAS_ASYNC_LDS 1
#else
#define HAS_ASYNC_LDS 0
#endif

#define ASGLOBAL __attribute__((address_space(1)))
#define ASLOCAL  __attribute__((address_space(3)))

// ----------------------------- helpers -------------------------------------

__device__ __forceinline__ void atomicMaxF32(float* addr, float val) {
    // monotone-bits trick: works with init = -INF (0xFF800000)
    if (!(val < 0.0f)) atomicMax((int*)addr, __float_as_int(val));
    else               atomicMin((unsigned int*)addr, __float_as_uint(val));
}

// ------------------------- pack / convert kernels --------------------------

__global__ void cvt_f32_bf16_kernel(const float* __restrict__ in,
                                    __bf16* __restrict__ out, size_t n) {
    size_t stride = (size_t)gridDim.x * blockDim.x;
    for (size_t i = (size_t)blockIdx.x * blockDim.x + threadIdx.x; i < n; i += stride)
        out[i] = (__bf16)in[i];
}

__global__ void bias_cat_kernel(const float* __restrict__ bl,
                                const float* __restrict__ br,
                                float* __restrict__ bcat, int half) {
    int i = blockIdx.x * blockDim.x + threadIdx.x;
    if (i < half)            bcat[i] = bl[i];
    else if (i < 2 * half)   bcat[i] = br[i - half];
}

__global__ void init_softmax_state_kernel(float* __restrict__ m,
                                          float* __restrict__ denom, size_t nmh,
                                          float* __restrict__ acc, size_t nacc) {
    size_t stride = (size_t)gridDim.x * blockDim.x;
    size_t i0 = (size_t)blockIdx.x * blockDim.x + threadIdx.x;
    for (size_t i = i0; i < nmh; i += stride) { m[i] = -INFINITY; denom[i] = 0.0f; }
    for (size_t i = i0; i < nacc; i += stride) acc[i] = 0.0f;
}

// --------------------------- WMMA bf16 GEMM --------------------------------
// C[M,Nc] = A[M,K] (row-major bf16) * W[Nc,K]^T (W row-major bf16) + bias[Nc]
//
// Block = 128 threads = 4 waves stacked along M (64 rows); each wave computes
// a 16x64 strip (4 accumulators) over the block's shared 64-column range.
// B tile (64x32 bf16) staged in LDS once per k-step (async global->LDS when
// available), shared by all 4 waves -> one A fragment feeds 4 WMMAs.
//
// Fragment layouts used here (contiguous k order, same for A and B):
//   A 16x32: lane L -> row M = L&15; k = 16*(L>=16) + e
//   B 32x16: lane L -> col N = L&15; k = 16*(L>=16) + e
//   C 16x16: lane L -> col N = L&15; VGPR v -> row M = v + 8*(L>=16)

__device__ __forceinline__ v16bf load_b_frag_lds(const __bf16* Bt, int n_local,
                                                 int lane) {
    const int col = n_local + (lane & 15);
    const int kb  = (lane >> 4) << 4;               // +16 for upper half-wave
    const __bf16* p = Bt + col * LDSROW + kb;       // 16B aligned
    v16bf b;
#pragma unroll
    for (int i = 0; i < 16; ++i) b[i] = p[i];       // 2x ds_load_b128
    return b;
}

template <bool RELU>
__global__ void wmma_gemm_bf16_kernel(const __bf16* __restrict__ A,
                                      const __bf16* __restrict__ W,
                                      const float* __restrict__ bias,
                                      float* __restrict__ Cmat,
                                      int M, int Nc, int K) {
    __shared__ __align__(16) __bf16 Btile[64 * LDSROW];

    const int lane  = threadIdx.x & 31;
    const int wave  = threadIdx.x >> 5;
    const int m0    = blockIdx.x * 64 + wave * 16;
    const int n0blk = blockIdx.y * 64;
    if (m0 >= M) return;                            // wave-uniform (never taken)

    // B-staging indices: thread copies 2x16B (rows crow and crow+32)
    const int crow = threadIdx.x >> 2;              // 0..31
    const int cch  = (threadIdx.x & 3) * 8;         // bf16 offset 0/8/16/24

    // A fragment addressing
    const int arow = m0 + (lane & 15);
    const int akb  = (lane >> 4) << 4;              // +16 for upper half-wave

    v8f acc[4] = {v8f{}, v8f{}, v8f{}, v8f{}};

    for (int k0 = 0; k0 < K; k0 += 32) {
        // ---- stage B tile 64x32 into LDS ----
        const __bf16* s0 = W + (size_t)(n0blk + crow) * K + k0 + cch;
        const __bf16* s1 = W + (size_t)(n0blk + crow + 32) * K + k0 + cch;
        __bf16* d0 = &Btile[crow * LDSROW + cch];
        __bf16* d1 = &Btile[(crow + 32) * LDSROW + cch];
#if HAS_ASYNC_LDS
        __builtin_amdgcn_global_load_async_to_lds_b128(
            (ASGLOBAL v4i*)s0, (ASLOCAL v4i*)d0, 0, 0);
        __builtin_amdgcn_global_load_async_to_lds_b128(
            (ASGLOBAL v4i*)s1, (ASLOCAL v4i*)d1, 0, 0);
#else
        *(uint4*)d0 = *(const uint4*)s0;
        *(uint4*)d1 = *(const uint4*)s1;
#endif

        // ---- A fragment direct from global (L2-resident) ----
        const __bf16* pa = A + (size_t)arow * K + k0 + akb;
        v16bf a;
#pragma unroll
        for (int i = 0; i < 8; ++i) { a[i] = pa[i]; a[i + 8] = pa[8 + i]; }
        __builtin_prefetch(pa + 32, 0, 1);

#if HAS_ASYNC_LDS
        __builtin_amdgcn_s_wait_asynccnt(0);
#endif
        __syncthreads();

        v16bf b0 = load_b_frag_lds(Btile, 0, lane);
        v16bf b1 = load_b_frag_lds(Btile, 16, lane);
        v16bf b2 = load_b_frag_lds(Btile, 32, lane);
        v16bf b3 = load_b_frag_lds(Btile, 48, lane);

        acc[0] = __builtin_amdgcn_wmma_f32_16x16x32_bf16(false, a, false, b0,
                                                         (short)0, acc[0], false, false);
        acc[1] = __builtin_amdgcn_wmma_f32_16x16x32_bf16(false, a, false, b1,
                                                         (short)0, acc[1], false, false);
        acc[2] = __builtin_amdgcn_wmma_f32_16x16x32_bf16(false, a, false, b2,
                                                         (short)0, acc[2], false, false);
        acc[3] = __builtin_amdgcn_wmma_f32_16x16x32_bf16(false, a, false, b3,
                                                         (short)0, acc[3], false, false);

        __syncthreads();                            // before B tile is rewritten
    }

    // ---- epilogue: bias (+ReLU) and store 4 tiles ----
    const int colb  = lane & 15;
    const int rbase = m0 + ((lane >> 4) << 3);
#pragma unroll
    for (int t = 0; t < 4; ++t) {
        const int col  = n0blk + t * 16 + colb;
        const float bs = bias[col];
#pragma unroll
        for (int v = 0; v < 8; ++v) {
            float r = acc[t][v] + bs;
            if (RELU) r = r > 0.0f ? r : 0.0f;
            Cmat[(size_t)(rbase + v) * Nc + col] = r;
        }
    }
}

// --------------------------- edge-phase kernels ----------------------------
// xlxr: [B*N][1024] f32, cols 0..511 = xl, 512..1023 = xr
// logits/p: [B][Etot][H] f32 ; m,denom: [B*N][H] ; acc: [B*N][H*C]

__device__ __forceinline__ void edge_endpoints(const long long* __restrict__ ei,
                                               int e, int E, int& src, int& dst) {
    if (e < E) { src = (int)ei[e]; dst = (int)ei[(size_t)E + e]; }
    else       { src = e - E; dst = e - E; }          // appended self loop
}

__global__ void edge_logits_kernel(const float* __restrict__ xlxr,
                                   const long long* __restrict__ ei,
                                   const float* __restrict__ att,   // [H*C]=512
                                   float* __restrict__ logits,
                                   float* __restrict__ m,
                                   int NN, int E, int Etot, int B) {
    const int lane = threadIdx.x & 31;
    const int gw = (blockIdx.x * blockDim.x + threadIdx.x) >> 5;    // global wave
    if (gw >= B * Etot) return;
    const int b = gw / Etot;
    const int e = gw - b * Etot;
    int src, dst;
    edge_endpoints(ei, e, E, src, dst);

    const float* pl = xlxr + ((size_t)b * NN + src) * 1024;         // xl[src]
    const float* pr = xlxr + ((size_t)b * NN + dst) * 1024 + 512;   // xr[dst]

    float acc[4] = {0.f, 0.f, 0.f, 0.f};
#pragma unroll
    for (int t = 0; t < 16; ++t) {                                  // 512 / 32
        const int idx = lane + t * 32;
        float v = pl[idx] + pr[idx];
        v = v > 0.0f ? v : NEG_SLOPE * v;
        acc[idx >> 7] += v * att[idx];
    }
#pragma unroll
    for (int off = 16; off > 0; off >>= 1) {
#pragma unroll
        for (int h = 0; h < 4; ++h) acc[h] += __shfl_down(acc[h], off, 32);
    }
    if (lane == 0) {
        float* pl4 = logits + ((size_t)b * Etot + e) * 4;
        float* pm  = m + ((size_t)b * NN + dst) * 4;
#pragma unroll
        for (int h = 0; h < 4; ++h) { pl4[h] = acc[h]; atomicMaxF32(&pm[h], acc[h]); }
    }
}

__global__ void edge_exp_kernel(const long long* __restrict__ ei,
                                float* __restrict__ logits,   // overwritten with p
                                const float* __restrict__ m,
                                float* __restrict__ denom,
                                int NN, int E, int Etot, int B) {
    const long long tid = (long long)blockIdx.x * blockDim.x + threadIdx.x;
    if (tid >= (long long)B * Etot * 4) return;
    const int h = (int)(tid & 3);
    const long long r = tid >> 2;
    const int b = (int)(r / Etot);
    const int e = (int)(r - (long long)b * Etot);
    int src, dst;
    edge_endpoints(ei, e, E, src, dst);
    (void)src;
    const size_t li = ((size_t)b * Etot + e) * 4 + h;
    const float p = __expf(logits[li] - m[((size_t)b * NN + dst) * 4 + h]);
    logits[li] = p;
    atomicAdd(&denom[((size_t)b * NN + dst) * 4 + h], p);
}

__global__ void edge_scatter_kernel(const float* __restrict__ xlxr,
                                    const long long* __restrict__ ei,
                                    const float* __restrict__ p,      // [B][Etot][4]
                                    const float* __restrict__ denom,
                                    float* __restrict__ acc,          // [B*N][512]
                                    int NN, int E, int Etot, int B) {
    const int lane = threadIdx.x & 31;
    const int gw = (blockIdx.x * blockDim.x + threadIdx.x) >> 5;
    if (gw >= B * Etot) return;
    const int b = gw / Etot;
    const int e = gw - b * Etot;
    int src, dst;
    edge_endpoints(ei, e, E, src, dst);

    const float* pp = p + ((size_t)b * Etot + e) * 4;
    const float* pd = denom + ((size_t)b * NN + dst) * 4;
    float alpha[4];
#pragma unroll
    for (int h = 0; h < 4; ++h) alpha[h] = pp[h] / pd[h];

    const float* xs = xlxr + ((size_t)b * NN + src) * 1024;   // xl[src]
    float* out = acc + ((size_t)b * NN + dst) * 512;
#pragma unroll
    for (int t = 0; t < 16; ++t) {
        const int idx = lane + t * 32;
        atomicAdd(&out[idx], xs[idx] * alpha[idx >> 7]);
    }
}

__global__ void head_mean_kernel(const float* __restrict__ acc,   // [rows][4][128]
                                 const float* __restrict__ gbias, // [128]
                                 __bf16* __restrict__ hmean,      // [rows][128] bf16
                                 size_t rows) {
    const size_t tid = (size_t)blockIdx.x * blockDim.x + threadIdx.x;
    if (tid >= rows * 128) return;
    const int c = (int)(tid & 127);
    const size_t row = tid >> 7;
    const float* a = acc + row * 512 + c;
    const float v = 0.25f * (a[0] + a[128] + a[256] + a[384]) + gbias[c];
    hmean[row * 128 + c] = (__bf16)v;
}

// ------------------------------ launcher -----------------------------------

extern "C" void kernel_launch(void* const* d_in, const int* in_sizes, int n_in,
                              void* d_out, int out_size, void* d_ws, size_t ws_size,
                              hipStream_t stream) {
    // problem sizes (reference: B=4, N=10000, E=160000, Cin=256, H=4, C=128)
    const int B = 4, NN = 10000, Cin = 256, H = 4, C = 128;
    const int HC = H * C;                       // 512
    const int E = in_sizes[1] / 2;              // 160000
    const int Etot = E + NN;                    // + self loops
    const int Mrows = B * NN;                   // 40000 (multiple of 64)

    const float*     x_nodes = (const float*)d_in[0];
    const long long* ei      = (const long long*)d_in[1];
    const float*     Wl      = (const float*)d_in[2];
    const float*     bl      = (const float*)d_in[3];
    const float*     Wr      = (const float*)d_in[4];
    const float*     br      = (const float*)d_in[5];
    const float*     att     = (const float*)d_in[6];
    const float*     gbias   = (const float*)d_in[7];
    const float*     fc_w    = (const float*)d_in[8];
    const float*     fc_b    = (const float*)d_in[9];
    float*           out     = (float*)d_out;

    // -------- workspace carve --------
    char* ws = (char*)d_ws;
    size_t off = 0;
    auto carve = [&](size_t bytes) { char* p = ws + off; off += (bytes + 255) & ~(size_t)255; return p; };
    __bf16* x_bf   = (__bf16*)carve((size_t)Mrows * Cin * sizeof(__bf16));
    __bf16* Wcat   = (__bf16*)carve((size_t)2 * HC * Cin * sizeof(__bf16));   // [1024][256]
    __bf16* fcw_bf = (__bf16*)carve((size_t)C * C * sizeof(__bf16));
    float*  bcat   = (float*)carve((size_t)2 * HC * sizeof(float));
    float*  xlxr   = (float*)carve((size_t)Mrows * 2 * HC * sizeof(float));   // [40000][1024]
    float*  logits = (float*)carve((size_t)B * Etot * H * sizeof(float));
    float*  mbuf   = (float*)carve((size_t)Mrows * H * sizeof(float));
    float*  denom  = (float*)carve((size_t)Mrows * H * sizeof(float));
    float*  accbuf = (float*)carve((size_t)Mrows * HC * sizeof(float));       // [40000][512]
    __bf16* hmean  = (__bf16*)carve((size_t)Mrows * C * sizeof(__bf16));

    // -------- stage 0: packing + state init --------
    {
        size_t n = (size_t)Mrows * Cin;
        cvt_f32_bf16_kernel<<<2048, 256, 0, stream>>>(x_nodes, x_bf, n);
        cvt_f32_bf16_kernel<<<512, 256, 0, stream>>>(Wl, Wcat, (size_t)HC * Cin);
        cvt_f32_bf16_kernel<<<512, 256, 0, stream>>>(Wr, Wcat + (size_t)HC * Cin,
                                                     (size_t)HC * Cin);
        cvt_f32_bf16_kernel<<<64, 256, 0, stream>>>(fc_w, fcw_bf, (size_t)C * C);
        bias_cat_kernel<<<(2 * HC + 255) / 256, 256, 0, stream>>>(bl, br, bcat, HC);
        init_softmax_state_kernel<<<4096, 256, 0, stream>>>(
            mbuf, denom, (size_t)Mrows * H, accbuf, (size_t)Mrows * HC);
    }

    // -------- stage 1: fused GEMM  [xl|xr] = x_bf * [Wl;Wr]^T + [bl;br] --------
    {
        dim3 grid(Mrows / 64, (2 * HC) / 64);   // 625 x 16
        wmma_gemm_bf16_kernel<false><<<grid, 128, 0, stream>>>(
            x_bf, Wcat, bcat, xlxr, Mrows, 2 * HC, Cin);
    }

    // -------- stage 2: edge attention (segment softmax over dst) --------
    {
        const long long waves = (long long)B * Etot;
        const int blocks = (int)((waves * 32 + 255) / 256);
        edge_logits_kernel<<<blocks, 256, 0, stream>>>(
            xlxr, ei, att, logits, mbuf, NN, E, Etot, B);

        const long long t2 = (long long)B * Etot * H;
        edge_exp_kernel<<<(int)((t2 + 255) / 256), 256, 0, stream>>>(
            ei, logits, mbuf, denom, NN, E, Etot, B);

        edge_scatter_kernel<<<blocks, 256, 0, stream>>>(
            xlxr, ei, logits, denom, accbuf, NN, E, Etot, B);
    }

    // -------- stage 3: head mean + gat_bias -> bf16, then fc GEMM + ReLU ------
    {
        const size_t t = (size_t)Mrows * C;
        head_mean_kernel<<<(int)((t + 255) / 256), 256, 0, stream>>>(
            accbuf, gbias, hmean, (size_t)Mrows);

        dim3 grid(Mrows / 64, C / 64);          // 625 x 2
        wmma_gemm_bf16_kernel<true><<<grid, 128, 0, stream>>>(
            hmean, fcw_bf, fc_b, out, Mrows, C, C);
    }
}
